// SyclInt8Linear_39874476376179
// MI455X (gfx1250) — compile-verified, matrix operand order checked by reading, not run on hardware
//
#include <hip/hip_runtime.h>

typedef __bf16 bf16;
typedef __attribute__((ext_vector_type(16))) __bf16 v16bf;
typedef __attribute__((ext_vector_type(8)))  __bf16 v8bf;
typedef __attribute__((ext_vector_type(8)))  float  v8f;
typedef __attribute__((ext_vector_type(4)))  float  v4f;
typedef __attribute__((ext_vector_type(4)))  int    v4i;

#define BM 128
#define BN 256
#define BK 32
#define LDR 40   // padded LDS row stride (80 B) to spread b128 reads over banks

// ---------------- prepass kernels: one-time dtype conversion into d_ws ------
__global__ __launch_bounds__(256)
void cvt_f32_to_bf16_k(const float* __restrict__ src, bf16* __restrict__ dst, long long n8)
{
    long long i = (long long)blockIdx.x * blockDim.x + threadIdx.x;
    if (i >= n8) return;
    const v4f* p = (const v4f*)src + i * 2;
    v4f a = p[0], b = p[1];
    v8bf o;
#pragma unroll
    for (int j = 0; j < 4; ++j) { o[j] = (bf16)a[j]; o[j + 4] = (bf16)b[j]; }
    ((v8bf*)dst)[i] = o;
}

__global__ __launch_bounds__(256)
void cvt_i32_to_bf16_k(const int* __restrict__ src, bf16* __restrict__ dst, long long n8)
{
    long long i = (long long)blockIdx.x * blockDim.x + threadIdx.x;
    if (i >= n8) return;
    const v4i* p = (const v4i*)src + i * 2;
    v4i a = p[0], b = p[1];
    v8bf o;
#pragma unroll
    for (int j = 0; j < 4; ++j) { o[j] = (bf16)(float)a[j]; o[j + 4] = (bf16)(float)b[j]; }
    ((v8bf*)dst)[i] = o;
}

// ---------------- main GEMM -------------------------------------------------
// y[m,n] = scale * sum_k x[m,k] * w[n,k] + bias[n]
// XBF/WBF: operand already bf16 in workspace (L2-resident) vs convert in-loop.
template <bool XBF, bool WBF>
__global__ __launch_bounds__(256, 1)
void int8_linear_wmma(const float* __restrict__ Xf, const bf16* __restrict__ Xb,
                      const int*   __restrict__ Wq, const bf16* __restrict__ Wb,
                      const float* __restrict__ scale, const float* __restrict__ bias,
                      float* __restrict__ Y, int M, int N, int K)
{
    __shared__ bf16 lA[2][BM * LDR];
    __shared__ bf16 lB[2][BN * LDR];

    const int tid   = threadIdx.x;
    const int wave  = tid >> 5;        // 0..7 (wave32)
    const int lane  = tid & 31;
    const int waveM = wave & 1;        // 2 waves along M  -> 64 rows each
    const int waveN = wave >> 1;       // 4 waves along N  -> 64 cols each
    const int g     = lane >> 4;       // half-wave group (K split in frags)
    const int r     = lane & 15;

    const int bn = blockIdx.x * BN;
    const int bm = blockIdx.y * BM;

    // Loader roles
    const int arow = tid >> 1;             // 0..127, half-row (16 K) of A
    const int akh  = (tid & 1) * 16;
    const int brow = tid;                  // 0..255, full row (32 K) of B
    const int aOff = arow * LDR + akh;
    const int bOff = brow * LDR;

    const float* xfPtr = Xf + (size_t)(bm + arow) * (size_t)K + akh;
    const bf16*  xbPtr = Xb + (size_t)(bm + arow) * (size_t)K + akh;
    const int*   wqPtr = Wq + (size_t)(bn + brow) * (size_t)K;
    const bf16*  wbPtr = Wb + (size_t)(bn + brow) * (size_t)K;

    v8f acc[4][4];
#pragma unroll
    for (int i = 0; i < 4; ++i)
#pragma unroll
        for (int j = 0; j < 4; ++j)
            acc[i][j] = (v8f)0.0f;

    struct StageRegs {
        v8bf ha[2];   // A bf16 path
        v8bf hb[4];   // B bf16 path
        v4f  fa[4];   // A f32 path
        v4i  iw[8];   // B i32 path
    };

    auto loadTiles = [&](int kt, StageRegs& s) {
        if (XBF) {
            const v8bf* p = (const v8bf*)(xbPtr + (size_t)kt * BK);
            s.ha[0] = p[0]; s.ha[1] = p[1];
        } else {
            const v4f* p = (const v4f*)(xfPtr + (size_t)kt * BK);
#pragma unroll
            for (int i = 0; i < 4; ++i) s.fa[i] = p[i];
        }
        if (WBF) {
            const v8bf* p = (const v8bf*)(wbPtr + (size_t)kt * BK);
#pragma unroll
            for (int i = 0; i < 4; ++i) s.hb[i] = p[i];
        } else {
            const v4i* p = (const v4i*)(wqPtr + (size_t)kt * BK);
#pragma unroll
            for (int i = 0; i < 8; ++i) s.iw[i] = p[i];
        }
    };

    auto stageTiles = [&](int buf, StageRegs& s) {
        v8bf pa0, pa1;
        if (XBF) { pa0 = s.ha[0]; pa1 = s.ha[1]; }
        else {
#pragma unroll
            for (int j = 0; j < 4; ++j) {
                pa0[j] = (bf16)s.fa[0][j]; pa0[j + 4] = (bf16)s.fa[1][j];
                pa1[j] = (bf16)s.fa[2][j]; pa1[j + 4] = (bf16)s.fa[3][j];
            }
        }
        *(v8bf*)(&lA[buf][aOff])     = pa0;
        *(v8bf*)(&lA[buf][aOff + 8]) = pa1;

        v8bf pb[4];
        if (WBF) {
#pragma unroll
            for (int i = 0; i < 4; ++i) pb[i] = s.hb[i];
        } else {
#pragma unroll
            for (int i = 0; i < 4; ++i)
#pragma unroll
                for (int j = 0; j < 4; ++j) {
                    pb[i][j]     = (bf16)(float)s.iw[2 * i][j];
                    pb[i][j + 4] = (bf16)(float)s.iw[2 * i + 1][j];
                }
        }
#pragma unroll
        for (int i = 0; i < 4; ++i)
            *(v8bf*)(&lB[buf][bOff + 8 * i]) = pb[i];
    };

    // Fragment layout per ISA 7.12.2 (16-bit A/B): lane group g holds
    // K = g*8..g*8+7 (low 8) and K = 16+g*8..16+g*8+7 (high 8).
    auto fragA = [&](int buf, int mt) -> v16bf {
        const int row = waveM * 64 + mt * 16 + r;
        const v8bf lo = *(const v8bf*)(&lA[buf][row * LDR + g * 8]);
        const v8bf hi = *(const v8bf*)(&lA[buf][row * LDR + 16 + g * 8]);
        return __builtin_shufflevector(lo, hi, 0,1,2,3,4,5,6,7,8,9,10,11,12,13,14,15);
    };
    auto fragB = [&](int buf, int nt) -> v16bf {
        const int row = waveN * 64 + nt * 16 + r;
        const v8bf lo = *(const v8bf*)(&lB[buf][row * LDR + g * 8]);
        const v8bf hi = *(const v8bf*)(&lB[buf][row * LDR + 16 + g * 8]);
        return __builtin_shufflevector(lo, hi, 0,1,2,3,4,5,6,7,8,9,10,11,12,13,14,15);
    };

    {   // prologue: stage tile 0
        StageRegs s;
        loadTiles(0, s);
        stageTiles(0, s);
    }
    __syncthreads();

    const int NKT = K / BK;
    for (int kt = 0; kt < NKT; ++kt) {
        const int buf = kt & 1;

        StageRegs s;
        const bool pre = (kt + 1 < NKT);
        if (pre) loadTiles(kt + 1, s);          // hide global/L2 latency
        if (kt + 2 < NKT) {                      // gfx1250 global_prefetch_b8
            if (XBF) __builtin_prefetch(xbPtr + (size_t)(kt + 2) * BK, 0, 1);
            else     __builtin_prefetch(xfPtr + (size_t)(kt + 2) * BK, 0, 1);
            if (WBF) __builtin_prefetch(wbPtr + (size_t)(kt + 2) * BK, 0, 1);
            else     __builtin_prefetch(wqPtr + (size_t)(kt + 2) * BK, 0, 1);
        }

        v16bf afr[4], bfr[4];
#pragma unroll
        for (int mt = 0; mt < 4; ++mt) afr[mt] = fragA(buf, mt);
#pragma unroll
        for (int nt = 0; nt < 4; ++nt) bfr[nt] = fragB(buf, nt);

#pragma unroll
        for (int mt = 0; mt < 4; ++mt)
#pragma unroll
            for (int nt = 0; nt < 4; ++nt)
                acc[mt][nt] = __builtin_amdgcn_wmma_f32_16x16x32_bf16(
                    false, afr[mt], false, bfr[nt],
                    (short)0, acc[mt][nt], false, false);

        if (pre) stageTiles(buf ^ 1, s);
        __syncthreads();
    }

    // Epilogue: y = acc * scale + bias
    // C/D layout: lane r = N col in tile; VGPR v = M row; lanes>=16 add M+8 (g*8).
    const float sc = scale[0];
#pragma unroll
    for (int nt = 0; nt < 4; ++nt) {
        const int n = bn + waveN * 64 + nt * 16 + r;
        const float bv = bias[n];
#pragma unroll
        for (int mt = 0; mt < 4; ++mt) {
            const int mbase = bm + waveM * 64 + mt * 16 + g * 8;
            float* out = Y + (size_t)mbase * (size_t)N + n;
#pragma unroll
            for (int v = 0; v < 8; ++v)
                out[(size_t)v * (size_t)N] = acc[mt][nt][v] * sc + bv;
        }
    }
}

extern "C" void kernel_launch(void* const* d_in, const int* in_sizes, int n_in,
                              void* d_out, int out_size, void* d_ws, size_t ws_size,
                              hipStream_t stream) {
    const float* X     = (const float*)d_in[0];
    const int*   Wq    = (const int*)  d_in[1];   // int8 values stored as int32
    const float* scale = (const float*)d_in[2];
    const float* bias  = (const float*)d_in[3];
    float*       Y     = (float*)d_out;

    const long long N  = in_sizes[3];                  // 11008
    const long long K  = (long long)in_sizes[1] / N;   // 4096
    const long long M  = (long long)in_sizes[0] / K;   // 8192
    const long long NK = N * K;
    const long long MK = M * K;

    const size_t needW    = (size_t)NK * 2;            // 90 MB  (L2-resident)
    const size_t needBoth = (size_t)(NK + MK) * 2;     // 157 MB (both L2-resident)

    const bool wbf = ws_size >= needW;
    const bool xbf = ws_size >= needBoth;

    bf16* Wb = (bf16*)d_ws;
    bf16* Xb = xbf ? (bf16*)((char*)d_ws + needW) : (bf16*)d_ws;

    if (wbf) {
        const long long n8 = NK / 8;
        cvt_i32_to_bf16_k<<<(unsigned)((n8 + 255) / 256), 256, 0, stream>>>(Wq, Wb, n8);
    }
    if (xbf) {
        const long long n8 = MK / 8;
        cvt_f32_to_bf16_k<<<(unsigned)((n8 + 255) / 256), 256, 0, stream>>>(X, Xb, n8);
    }

    dim3 grid((unsigned)(N / BN), (unsigned)(M / BM));
    if (xbf && wbf)
        int8_linear_wmma<true,  true ><<<grid, 256, 0, stream>>>(X, Xb, Wq, Wb, scale, bias, Y, (int)M, (int)N, (int)K);
    else if (wbf)
        int8_linear_wmma<false, true ><<<grid, 256, 0, stream>>>(X, Xb, Wq, Wb, scale, bias, Y, (int)M, (int)N, (int)K);
    else
        int8_linear_wmma<false, false><<<grid, 256, 0, stream>>>(X, Xb, Wq, Wb, scale, bias, Y, (int)M, (int)N, (int)K);
}
